// MultiQueryAttention_53386443489405
// MI455X (gfx1250) — matrix operand order repaired and checked by reading, not compiled, not run on hardware
//
#include <hip/hip_runtime.h>
#include <hip/hip_bf16.h>

// ---------------------------------------------------------------------------
// MQA on gfx1250 (MI455X), compute-bound (~283 GFLOP vs ~0.4 GB traffic).
// All matmul FLOPs routed through v_wmma_f32_16x16x32_bf16 (wave32 WMMA).
// K-tile staging in the attention kernel uses the Tensor Data Mover
// (tensor_load_to_lds + s_wait_tensorcnt) when the builtin is available.
// ---------------------------------------------------------------------------

constexpr int B  = 4;
constexpr int S  = 2048;
constexpr int D  = 2048;
constexpr int H  = 16;
constexpr int DH = 128;           // head dim

typedef __attribute__((ext_vector_type(16))) __bf16 v16bf;
typedef __attribute__((ext_vector_type(8)))  float  v8f;
typedef __attribute__((ext_vector_type(4)))  unsigned u32x4;
typedef __attribute__((ext_vector_type(8)))  int      i32x8;
typedef __attribute__((ext_vector_type(4)))  int      i32x4;

#if __has_builtin(__builtin_amdgcn_tensor_load_to_lds) && \
    __has_builtin(__builtin_amdgcn_s_wait_tensorcnt)
#define USE_TDM 1
#else
#define USE_TDM 0
#endif

// ---------------------------------------------------------------------------
// Fragment loaders.  WMMA 16x16x32 bf16 VGPR layouts (ISA 7.12.2):
//  A (16x32, MxK): lane m = lane&15; half = lane>>4 selects K-halves;
//    VGPR i (i<4): K = 2i,2i+1 (+8 if half); VGPR i>=4: K = 16+2(i-4),.. (+8 if half)
//  B (32x16, KxN): lane n = lane&15; half = lane>>4; same K order as A:
//    VGPR j (j<4): K = 2j,2j+1 (+8 if half); VGPR j>=4: K = 16+2(j-4),.. (+8 if half)
//  C/D (16x16 f32): VGPR r holds row M = r + (half?8:0), column N = lane&15.
// B-operand tiles are staged TRANSPOSED in LDS ([n][k]) so each lane's frag is
// two runs of 4 contiguous dwords (vectorizes to ds_load_b128).
// ---------------------------------------------------------------------------

__device__ inline v16bf load_frag_a(const __bf16* base, int row0, int ldr,
                                    int k0, int lane) {
  const int m = lane & 15, half = lane >> 4;
  union { v16bf v; unsigned u[8]; } f;
  const __bf16* rp = base + (size_t)(row0 + m) * ldr + k0 + half * 8;
#pragma unroll
  for (int i = 0; i < 8; ++i) {
    const int kk = (i < 4) ? (2 * i) : (16 + 2 * (i - 4));
    f.u[i] = *(const unsigned*)(rp + kk);          // 2 packed bf16 per dword
  }
  return f.v;
}

// base laid out [n][k] (transposed), ldr = k-stride of that layout
__device__ inline v16bf load_frag_bt(const __bf16* base, int n0, int ldr,
                                     int k0, int lane) {
  const int n = lane & 15, half = lane >> 4;
  union { v16bf v; unsigned u[8]; } f;
  const __bf16* rp = base + (size_t)(n0 + n) * ldr + k0 + half * 8;
#pragma unroll
  for (int i = 0; i < 8; ++i)
    f.u[i] = *(const unsigned*)(rp + ((i < 4) ? (2 * i) : (16 + 2 * (i - 4))));  // same k order as A
  return f.v;
}

// ---------------------------------------------------------------------------
// Generic tiled GEMM + bias:  C[M,N] = A[M,K] * W[K,N] + bias[N]
// A: f32 or bf16 (converted to bf16 while staging); W: f32 -> bf16 in LDS.
// Block = 256 threads = 8 waves in a 2(M) x 4(N) grid; each wave owns a 64x32
// sub-tile = 4x2 WMMA accumulators. Workgroup tile 128x128, K-step 32.
// ---------------------------------------------------------------------------
template <typename TIN, typename TOUT>
__global__ __launch_bounds__(256) void gemm_bias_kernel(
    const TIN* __restrict__ A, const float* __restrict__ W,
    const float* __restrict__ bias, TOUT* __restrict__ C,
    int M, int N, int K) {
  constexpr int BM = 128, BN = 128, BK = 32;
  __shared__ alignas(16) __bf16 sA[BM * BK];       // [m][k]
  __shared__ alignas(16) __bf16 sBt[BN * BK];      // [n][k] (transposed W tile)

  const int tid  = threadIdx.x;
  const int lane = tid & 31;
  const int wave = tid >> 5;
  const int wm   = wave >> 2;                      // 0..1
  const int wn   = wave & 3;                       // 0..3
  const int bm0  = blockIdx.y * BM;
  const int bn0  = blockIdx.x * BN;

  v8f acc[4][2] = {};

  for (int k0 = 0; k0 < K; k0 += BK) {
    // stage A tile (BM x BK), convert to bf16
#pragma unroll
    for (int i = tid * 4; i < BM * BK; i += 256 * 4) {
      const int r = i / BK, c = i % BK;
      const TIN* src = A + (size_t)(bm0 + r) * K + k0 + c;
#pragma unroll
      for (int j = 0; j < 4; ++j)
        sA[r * BK + c + j] = (__bf16)((float)src[j]);
    }
    // stage W tile (BK x BN), transposed into [n][k]
#pragma unroll
    for (int i = tid * 4; i < BK * BN; i += 256 * 4) {
      const int r = i / BN, c = i % BN;
      const float* src = W + (size_t)(k0 + r) * N + bn0 + c;
#pragma unroll
      for (int j = 0; j < 4; ++j)
        sBt[(c + j) * BK + r] = (__bf16)src[j];
    }
    if (k0 + BK < K)
      __builtin_prefetch(A + (size_t)(bm0 + (tid >> 1)) * K + k0 + BK, 0, 1);
    __syncthreads();

    v16bf bf[2];
#pragma unroll
    for (int nt = 0; nt < 2; ++nt)
      bf[nt] = load_frag_bt(sBt, wn * 32 + nt * 16, BK, 0, lane);
#pragma unroll
    for (int mt = 0; mt < 4; ++mt) {
      const v16bf af = load_frag_a(sA, wm * 64 + mt * 16, BK, 0, lane);
#pragma unroll
      for (int nt = 0; nt < 2; ++nt)
        acc[mt][nt] = __builtin_amdgcn_wmma_f32_16x16x32_bf16(
            false, af, false, bf[nt], (short)0, acc[mt][nt], false, false);
    }
    __syncthreads();
  }

  // epilogue: bias + store (C layout: row = r + half*8, col = lane&15)
  const int half = lane >> 4, nn = lane & 15;
#pragma unroll
  for (int mt = 0; mt < 4; ++mt)
#pragma unroll
    for (int nt = 0; nt < 2; ++nt)
#pragma unroll
      for (int r = 0; r < 8; ++r) {
        const int m = bm0 + wm * 64 + mt * 16 + r + half * 8;
        const int n = bn0 + wn * 32 + nt * 16 + nn;
        const float v = acc[mt][nt][r] + bias[n];
        C[(size_t)m * N + n] = (TOUT)v;
      }
}

// ---------------------------------------------------------------------------
// Flash attention (multi-query: one K/V head shared by all 16 Q heads).
// grid = (S/128, H, B), block = 256 (8 waves).  Each wave owns 16 query rows
// with q-fragments resident in registers (4 x v16bf covering DH=128).
// K blocks of 64 keys staged to LDS: K as [key][d] (via TDM when available),
// V transposed as [d][key] via VALU staging.
// Online softmax: row stats live per accumulator-register index r (rows are
// uniform across a 16-lane half), reductions via __shfl_xor 8/4/2/1.
// P (C-layout) -> A-layout conversion via per-wave LDS staging buffer.
// ---------------------------------------------------------------------------
__global__ __launch_bounds__(256) void mqa_flash_kernel(
    const __bf16* __restrict__ qh,   // [B,S,D]  head h at cols [h*DH, h*DH+DH)
    const __bf16* __restrict__ kh,   // [B,S,DH]
    const __bf16* __restrict__ vh,   // [B,S,DH]
    __bf16* __restrict__ ao) {       // [B,S,D]
  constexpr int KB = 64;             // keys per block iteration
  __shared__ alignas(16) __bf16 sK[KB * DH];        // [key][d]
  __shared__ alignas(16) __bf16 sVt[DH * KB];       // [d][key]  (transposed)
  __shared__ alignas(16) __bf16 sP[8 * 16 * KB];    // per-wave P staging

  const int tid = threadIdx.x, lane = tid & 31, wave = tid >> 5;
  const int b = blockIdx.z, h = blockIdx.y;
  const int q0 = blockIdx.x * 128 + wave * 16;
  const int ln = lane & 15, half = lane >> 4;
  const float scale = 0.08838834764831845f;         // 1/sqrt(DH)

  // q fragments (A layout), pre-scaled by 1/sqrt(DH)
  v16bf qf[4];
  const __bf16* qrow = qh + ((size_t)b * S + q0 + ln) * D + h * DH;
#pragma unroll
  for (int kc = 0; kc < 4; ++kc) {
#pragma unroll
    for (int i = 0; i < 8; ++i) {
      const int kb = ((i < 4) ? 2 * i : 16 + 2 * (i - 4)) + half * 8;
      const __bf16* p = qrow + kc * 32 + kb;
      qf[kc][2 * i]     = (__bf16)((float)p[0] * scale);
      qf[kc][2 * i + 1] = (__bf16)((float)p[1] * scale);
    }
  }

  v8f o[8] = {};                                    // 16 x 128 output accum
  float rowM[8], rowL[8];
#pragma unroll
  for (int r = 0; r < 8; ++r) { rowM[r] = -1e30f; rowL[r] = 0.0f; }

  const __bf16* kbase = kh + (size_t)b * S * DH;
  const __bf16* vbase = vh + (size_t)b * S * DH;
  __bf16* myP = sP + wave * 16 * KB;

  for (int kb0 = 0; kb0 < S; kb0 += KB) {
    // ---- stage K tile [64 x 128] ----
#if USE_TDM
    // Tensor Data Mover: 2D tile [KB rows x DH cols] of the [S x DH] bf16
    // tensor -> LDS sK.  D# packing per CDNA5 ISA section 8.3/8.4.
    if (wave == 0) {
      const unsigned long long ga =
          (unsigned long long)(const void*)(kbase + (size_t)kb0 * DH);
      const unsigned ldsoff =
          (unsigned)(unsigned long long)(const void*)sK;  // addr[31:0] = LDS off
      u32x4 g0;
      g0[0] = 1u;                                         // count=1, user D#
      g0[1] = ldsoff;                                     // lds_addr [63:32]
      g0[2] = (unsigned)(ga & 0xFFFFFFFFull);             // global_addr lo
      g0[3] = (unsigned)((ga >> 32) & 0x01FFFFFFull)      // global_addr hi
              | (2u << 30);                               // type=2 ("image")
      i32x8 g1;
      g1[0] = 0x00010000;                  // wg_mask=0, data_size=1 (2 bytes)
      g1[1] = (int)((unsigned)DH << 16);   // atomic_bar=0 | tensor_dim0 lo16
      g1[2] = (int)((unsigned)S << 16);    // tensor_dim0 hi16 | tensor_dim1 lo16
      g1[3] = (int)((unsigned)DH << 16);   // tensor_dim1 hi16 | tile_dim0=128
      g1[4] = KB;                          // tile_dim1=64 | tile_dim2=0
      g1[5] = DH;                          // tensor_dim0_stride lo32
      g1[6] = 0;                           // stride0 hi16 | stride1 lo16
      g1[7] = 0;                           // stride1 hi32
      const i32x4 z4 = {0, 0, 0, 0};       // groups 2/3 unused (2D tensor)
      const i32x8 z8 = {0, 0, 0, 0, 0, 0, 0, 0};
      __builtin_amdgcn_tensor_load_to_lds(g0, g1, z4, z4, z8, 0);
      __builtin_amdgcn_s_wait_tensorcnt(0);
    }
#else
#pragma unroll
    for (int i = tid * 4; i < KB * DH; i += 256 * 4) {
      const int r = i / DH, c = i % DH;
      const __bf16* srck = kbase + (size_t)(kb0 + r) * DH + c;
#pragma unroll
      for (int j = 0; j < 4; ++j) sK[r * DH + c + j] = srck[j];
    }
#endif

    // ---- stage V tile transposed [d][key] ----
#pragma unroll
    for (int i = tid * 4; i < KB * DH; i += 256 * 4) {
      const int r = i / DH, c = i % DH;
      const __bf16* srcv = vbase + (size_t)(kb0 + r) * DH + c;
#pragma unroll
      for (int j = 0; j < 4; ++j) sVt[(c + j) * KB + r] = srcv[j];
    }
    if (kb0 + KB < S) {
      __builtin_prefetch(kbase + (size_t)(kb0 + KB) * DH + (tid << 5), 0, 1);
      __builtin_prefetch(vbase + (size_t)(kb0 + KB) * DH + (tid << 5), 0, 1);
    }
    __syncthreads();

    // ---- scores S = (q * 1/sqrt(DH)) @ K^T  : 4 n-tiles x 4 k-chunks ----
    v8f sacc[4] = {};
#pragma unroll
    for (int nt = 0; nt < 4; ++nt)
#pragma unroll
      for (int kc = 0; kc < 4; ++kc) {
        const v16bf bk = load_frag_bt(sK, nt * 16, DH, kc * 32, lane);
        sacc[nt] = __builtin_amdgcn_wmma_f32_16x16x32_bf16(
            false, qf[kc], false, bk, (short)0, sacc[nt], false, false);
      }

    // ---- online softmax update (per accumulator row r) ----
#pragma unroll
    for (int r = 0; r < 8; ++r) {
      float bm = fmaxf(fmaxf(sacc[0][r], sacc[1][r]),
                       fmaxf(sacc[2][r], sacc[3][r]));
#pragma unroll
      for (int off = 8; off; off >>= 1)
        bm = fmaxf(bm, __shfl_xor(bm, off, 32));    // reduce 16-lane half
      const float mNew  = fmaxf(rowM[r], bm);
      const float alpha = __expf(rowM[r] - mNew);
      rowM[r] = mNew;
      float ps = 0.0f;
#pragma unroll
      for (int nt = 0; nt < 4; ++nt) {
        const float p = __expf(sacc[nt][r] - mNew);
        sacc[nt][r] = p;
        ps += p;
      }
#pragma unroll
      for (int off = 8; off; off >>= 1)
        ps += __shfl_xor(ps, off, 32);
      rowL[r] = rowL[r] * alpha + ps;
#pragma unroll
      for (int dt = 0; dt < 8; ++dt)
        o[dt][r] *= alpha;
    }

    // ---- P: C-layout -> LDS -> A-layout fragments ----
#pragma unroll
    for (int nt = 0; nt < 4; ++nt)
#pragma unroll
      for (int r = 0; r < 8; ++r)
        myP[(r + half * 8) * KB + nt * 16 + ln] = (__bf16)sacc[nt][r];

    v16bf pf[2];
#pragma unroll
    for (int j = 0; j < 2; ++j)
      pf[j] = load_frag_a(myP, 0, KB, j * 32, lane);

    // ---- O += P @ V  : 8 d-tiles x 2 k-chunks ----
#pragma unroll
    for (int dt = 0; dt < 8; ++dt)
#pragma unroll
      for (int j = 0; j < 2; ++j) {
        const v16bf bv = load_frag_bt(sVt, dt * 16, KB, j * 32, lane);
        o[dt] = __builtin_amdgcn_wmma_f32_16x16x32_bf16(
            false, pf[j], false, bv, (short)0, o[dt], false, false);
      }
    __syncthreads();                                // protect K/V tiles
  }

  // ---- normalize and write [B,S,D] bf16 ----
#pragma unroll
  for (int dt = 0; dt < 8; ++dt)
#pragma unroll
    for (int r = 0; r < 8; ++r) {
      const float val = o[dt][r] / rowL[r];
      const int row = q0 + r + half * 8;
      ao[((size_t)b * S + row) * D + h * DH + dt * 16 + ln] = (__bf16)val;
    }
}

// ---------------------------------------------------------------------------
extern "C" void kernel_launch(void* const* d_in, const int* in_sizes, int n_in,
                              void* d_out, int out_size, void* d_ws,
                              size_t ws_size, hipStream_t stream) {
  (void)in_sizes; (void)n_in; (void)out_size; (void)ws_size;
  const float* q  = (const float*)d_in[0];
  const float* k  = (const float*)d_in[1];
  const float* v  = (const float*)d_in[2];
  const float* Wq = (const float*)d_in[3];
  const float* bq = (const float*)d_in[4];
  const float* Wk = (const float*)d_in[5];
  const float* bk = (const float*)d_in[6];
  const float* Wv = (const float*)d_in[7];
  const float* bv = (const float*)d_in[8];
  const float* Wo = (const float*)d_in[9];
  const float* bo = (const float*)d_in[10];
  float* out = (float*)d_out;

  // workspace: bf16 intermediates (qh 32MB, kh/vh 2MB each, attn_out 32MB)
  __bf16* qh = (__bf16*)d_ws;
  __bf16* kh = qh + (size_t)B * S * D;
  __bf16* vh = kh + (size_t)B * S * DH;
  __bf16* ao = vh + (size_t)B * S * DH;

  const dim3 blk(256);
  const int M = B * S;                              // 8192

  // Projections (f32 in, bf16 out)
  gemm_bias_kernel<float, __bf16>
      <<<dim3(D / 128, M / 128), blk, 0, stream>>>(q, Wq, bq, qh, M, D, D);
  gemm_bias_kernel<float, __bf16>
      <<<dim3(DH / 128, M / 128), blk, 0, stream>>>(k, Wk, bk, kh, M, DH, D);
  gemm_bias_kernel<float, __bf16>
      <<<dim3(DH / 128, M / 128), blk, 0, stream>>>(v, Wv, bv, vh, M, DH, D);

  // Flash attention over shared K/V head
  mqa_flash_kernel<<<dim3(S / 128, H, B), blk, 0, stream>>>(qh, kh, vh, ao);

  // Output projection (bf16 in, f32 out)
  gemm_bias_kernel<__bf16, float>
      <<<dim3(D / 128, M / 128), blk, 0, stream>>>(ao, Wo, bo, out, M, D, D);
}
